// Model_39676907880777
// MI455X (gfx1250) — matrix operand order repaired and checked by reading, not compiled, not run on hardware
//
#include <hip/hip_runtime.h>
#include <hip/hip_bf16.h>

typedef __attribute__((ext_vector_type(16))) _Float16 v16h;
typedef __attribute__((ext_vector_type(8)))  _Float16 v8h;
typedef __attribute__((ext_vector_type(8)))  float    v8f;

#define LQ 4096
#define LK 4096
#define DH 128
#define BM 128         // q rows per block (32 per wave x 4 waves)
#define BN 64          // k columns per iteration
#define RT 2           // 16-row tiles per wave

#define KT_STRIDE 136  // 272B rows -> 68-dword stride -> conflict-free b128 frag loads
#define VT_STRIDE 72   // 144B rows (16B aligned)
#define P_STRIDE  72   // 144B rows (16B aligned)

__launch_bounds__(128)
__global__ void fa_fwd_kernel(const float* __restrict__ Q,
                              const float* __restrict__ K,
                              const float* __restrict__ V,
                              float* __restrict__ Out)
{
    __shared__ _Float16 sK[64  * KT_STRIDE];            // K tile, row-major [k][d]
    __shared__ _Float16 sV[128 * VT_STRIDE];            // V tile, transposed [d][k]
    __shared__ _Float16 sP[4 * 32 * P_STRIDE];          // per-wave P buffer [q][k]

    const int tid  = threadIdx.x;
    const int wave = tid >> 5;
    const int lane = tid & 31;
    const int l16  = lane & 15;
    const int hf   = lane >> 4;          // lane half (0: lanes 0-15, 1: lanes 16-31)

    const int b  = blockIdx.y;
    const int q0 = blockIdx.x * BM + wave * (16 * RT);  // first q row of this wave

    const float qscale = 1.0f / 11.313708498984761f;    // 1/sqrt(128)

    // ---- Load Q fragments in WMMA 16-bit A-layout (16x32 per fragment, 4 per row-tile)
    v16h qa[RT][4];
    #pragma unroll
    for (int rt = 0; rt < RT; ++rt) {
        const float* qrow = Q + ((size_t)b * LQ + q0 + rt * 16 + l16) * DH;
        #pragma unroll
        for (int ds = 0; ds < 4; ++ds) {
            const int d0 = ds * 32 + 8 * hf;
            v16h a;
            #pragma unroll
            for (int j = 0; j < 8; ++j) a[j]     = (_Float16)(qrow[d0 + j]      * qscale);
            #pragma unroll
            for (int j = 0; j < 8; ++j) a[j + 8] = (_Float16)(qrow[d0 + 16 + j] * qscale);
            qa[rt][ds] = a;
        }
    }

    // ---- accumulators / softmax state. Row per lane: r = rt*16 + v + 8*hf (C-layout)
    v8f   o[RT][8] = {};
    float m[RT][8], l[RT][8];
    #pragma unroll
    for (int rt = 0; rt < RT; ++rt)
        #pragma unroll
        for (int v = 0; v < 8; ++v) { m[rt][v] = -3.0e38f; l[rt][v] = 0.0f; }

    _Float16* pw = &sP[wave * 32 * P_STRIDE];

    for (int kb = 0; kb < LK; kb += BN) {
        __syncthreads();

        // ---- stage K tile: 64x128 f32 -> f16 LDS row-major (packed b64 stores)
        const float* kbase = K + ((size_t)b * LK + kb) * DH;
        for (int i = tid; i < 64 * 32; i += 128) {
            const int r  = i >> 5;
            const int c4 = (i & 31) * 4;
            float4 x = *(const float4*)(kbase + r * DH + c4);
            union { _Float16 h[4]; uint2 u; } pk;
            pk.h[0] = (_Float16)x.x; pk.h[1] = (_Float16)x.y;
            pk.h[2] = (_Float16)x.z; pk.h[3] = (_Float16)x.w;
            *(uint2*)&sK[r * KT_STRIDE + c4] = pk.u;
        }
        // ---- stage V tile transposed: V[k][d] -> sV[d][k] (f16)
        const float* vbase = V + ((size_t)b * LK + kb) * DH;
        for (int i = tid; i < 64 * 32; i += 128) {
            const int r  = i >> 5;           // k within tile
            const int c4 = (i & 31) * 4;     // d
            float4 x = *(const float4*)(vbase + r * DH + c4);
            sV[(c4 + 0) * VT_STRIDE + r] = (_Float16)x.x;
            sV[(c4 + 1) * VT_STRIDE + r] = (_Float16)x.y;
            sV[(c4 + 2) * VT_STRIDE + r] = (_Float16)x.z;
            sV[(c4 + 3) * VT_STRIDE + r] = (_Float16)x.w;
        }
        __syncthreads();

        // ---- prefetch next K/V tile into cache while we compute on this one
        if (kb + BN < LK) {
            const char* kn = (const char*)(kbase + (size_t)BN * DH);
            const char* vn = (const char*)(vbase + (size_t)BN * DH);
            // tile = 32KB = 256 cachelines of 128B; 128 threads x 2 lines each
            __builtin_prefetch(kn + tid * 256,       0, 1);
            __builtin_prefetch(kn + tid * 256 + 128, 0, 1);
            __builtin_prefetch(vn + tid * 256,       0, 1);
            __builtin_prefetch(vn + tid * 256 + 128, 0, 1);
        }

        // ---- S = Q @ K^T : 4 N-tiles x 2 row-tiles; each B-frag feeds both row-tiles
        v8f s[RT][4];
        #pragma unroll
        for (int t = 0; t < 4; ++t) {
            v8f a0 = {}, a1 = {};
            #pragma unroll
            for (int ds = 0; ds < 4; ++ds) {
                const _Float16* bp = &sK[(t * 16 + l16) * KT_STRIDE + ds * 32 + 8 * hf];
                union { v16h v; v8h h[2]; } bu;
                bu.h[0] = *(const v8h*)(bp);
                bu.h[1] = *(const v8h*)(bp + 16);
                a0 = __builtin_amdgcn_wmma_f32_16x16x32_f16(
                        false, qa[0][ds], false, bu.v, (short)0, a0, false, false);
                a1 = __builtin_amdgcn_wmma_f32_16x16x32_f16(
                        false, qa[1][ds], false, bu.v, (short)0, a1, false, false);
            }
            s[0][t] = a0;
            s[1][t] = a1;
        }

        // ---- online softmax per row-tile
        #pragma unroll
        for (int rt = 0; rt < RT; ++rt) {
            #pragma unroll
            for (int v = 0; v < 8; ++v) {
                float rm = fmaxf(fmaxf(s[rt][0][v], s[rt][1][v]),
                                 fmaxf(s[rt][2][v], s[rt][3][v]));
                #pragma unroll
                for (int off = 1; off < 16; off <<= 1)
                    rm = fmaxf(rm, __shfl_xor(rm, off, 32));
                const float mn = fmaxf(m[rt][v], rm);
                const float sc = __expf(m[rt][v] - mn);
                m[rt][v] = mn;
                float rs = 0.0f;
                #pragma unroll
                for (int t = 0; t < 4; ++t) {
                    const float p = __expf(s[rt][t][v] - mn);
                    s[rt][t][v] = p;
                    rs += p;
                }
                #pragma unroll
                for (int off = 1; off < 16; off <<= 1)
                    rs += __shfl_xor(rs, off, 32);
                l[rt][v] = l[rt][v] * sc + rs;
                #pragma unroll
                for (int n = 0; n < 8; ++n) o[rt][n][v] *= sc;
            }
        }

        // ---- transpose P (C-layout -> A-layout) through per-wave LDS
        #pragma unroll
        for (int rt = 0; rt < RT; ++rt)
            #pragma unroll
            for (int t = 0; t < 4; ++t)
                #pragma unroll
                for (int v = 0; v < 8; ++v)
                    pw[(rt * 16 + v + 8 * hf) * P_STRIDE + t * 16 + l16] =
                        (_Float16)s[rt][t][v];
        // LDS ops are in-order within a wave; block compiler reordering + drain DS counter
        asm volatile("s_wait_dscnt 0" ::: "memory");

        // ---- O += P @ V : 2 k-steps of 32, 8 d-tiles; each V B-frag feeds both row-tiles
        #pragma unroll
        for (int kc = 0; kc < 2; ++kc) {
            union { v16h v; v8h h[2]; } au[RT];
            #pragma unroll
            for (int rt = 0; rt < RT; ++rt) {
                const _Float16* ap = &pw[(rt * 16 + l16) * P_STRIDE + kc * 32 + 8 * hf];
                au[rt].h[0] = *(const v8h*)(ap);        // k = kc*32 + 8*hf .. +7
                au[rt].h[1] = *(const v8h*)(ap + 16);   // k = kc*32 + 16 + 8*hf .. +7
            }
            #pragma unroll
            for (int n = 0; n < 8; ++n) {
                const _Float16* bp = &sV[(n * 16 + l16) * VT_STRIDE + kc * 32 + 8 * hf];
                union { v16h v; v8h h[2]; } bu;
                bu.h[0] = *(const v8h*)(bp);
                bu.h[1] = *(const v8h*)(bp + 16);
                o[0][n] = __builtin_amdgcn_wmma_f32_16x16x32_f16(
                            false, au[0].v, false, bu.v, (short)0, o[0][n], false, false);
                o[1][n] = __builtin_amdgcn_wmma_f32_16x16x32_f16(
                            false, au[1].v, false, bu.v, (short)0, o[1][n], false, false);
            }
        }
    }

    // ---- normalize and write O (fp32)
    #pragma unroll
    for (int rt = 0; rt < RT; ++rt) {
        float* orow = Out + ((size_t)b * LQ + q0 + rt * 16) * DH;
        #pragma unroll
        for (int v = 0; v < 8; ++v) {
            const float inv = 1.0f / l[rt][v];
            const int r = v + 8 * hf;
            #pragma unroll
            for (int n = 0; n < 8; ++n)
                orow[r * DH + n * 16 + l16] = o[rt][n][v] * inv;
        }
    }
}

extern "C" void kernel_launch(void* const* d_in, const int* in_sizes, int n_in,
                              void* d_out, int out_size, void* d_ws, size_t ws_size,
                              hipStream_t stream) {
    const float* Q = (const float*)d_in[0];
    const float* K = (const float*)d_in[1];
    const float* V = (const float*)d_in[2];
    float* Out = (float*)d_out;

    const int nb = in_sizes[0] / (LQ * DH);   // batch count (16)
    dim3 grid(LQ / BM, nb);
    fa_fwd_kernel<<<grid, 128, 0, stream>>>(Q, K, V, Out);
}
